// GATBlock_4372276707492
// MI455X (gfx1250) — matrix operand order repaired and checked by reading, not compiled, not run on hardware
//
#include <hip/hip_runtime.h>
#include <hip/hip_bf16.h>

// GATv2 block for MI455X (gfx1250, wave32, WMMA).
// All GEMMs run through v_wmma_f32_16x16x32_bf16; epilogues fused.
// Workspace usage ~132 MB (v, xl, xr, logits/ex, node max/den, accum, bf16 weights).

typedef __bf16 bf16_t;
typedef __attribute__((ext_vector_type(16))) __bf16 v16bf;
typedef __attribute__((ext_vector_type(8)))  float  v8f;

#define DD   128   // node feature dim
#define HIDD 512   // FFN hidden
#define EDD  32    // edge feature dim
#define HH   8     // heads
#define CC   16    // channels per head

// ---------- helpers ----------

__device__ __forceinline__ bf16_t f2bf(float f) {
  // round-to-nearest-even float -> bf16 (bit-level; avoids reliance on fp cast support)
  unsigned u = __float_as_uint(f);
  u += 0x7FFFu + ((u >> 16) & 1u);
  unsigned short s = (unsigned short)(u >> 16);
  return __builtin_bit_cast(bf16_t, s);
}

// A fragment: 16x32 bf16 tile, row-major source with leading dim `ld` (elements).
// Contiguous k order used here (A and B match): lanes 0-15 -> rows 0-15, K 0..15; lanes 16-31 same rows,
// K 16..31.
__device__ __forceinline__ v16bf load_A_frag(const bf16_t* tile, int ld, int k0, int lane) {
  const int row  = lane & 15;
  const int half = lane >> 4;
  const bf16_t* p = tile + row * ld + k0 + half * 16;
  v16bf a;
#pragma unroll
  for (int j = 0; j < 8; ++j) a[j] = p[j];
#pragma unroll
  for (int j = 0; j < 8; ++j) a[8 + j] = p[8 + j];
  return a;
}

// B fragment: 32x16 bf16 tile taken from a transposed weight wT[N][K] (K contiguous).
// Contiguous k order used here (matches A): lane L -> column L%16; K block = (L/16)*16 + j, j=0..15 (contiguous 32B).
__device__ __forceinline__ v16bf load_B_frag(const bf16_t* wT, int Kdim, int n0, int k0, int lane) {
  const int col = lane & 15;
  const int kq  = (lane >> 4) * 16;
  const bf16_t* p = wT + (size_t)(n0 + col) * Kdim + k0 + kq;
  v16bf b;
#pragma unroll
  for (int j = 0; j < 16; ++j) b[j] = p[j];
  return b;
}

__device__ __forceinline__ v8f wmma_bf16(v16bf a, v16bf b, v8f c) {
  // (neg_a, A, neg_b, B, c_mod, C, reuse_a, reuse_b)
  return __builtin_amdgcn_wmma_f32_16x16x32_bf16(false, a, false, b, (short)0, c, false, false);
}

// ---------- K0: convert fp32 weight [K][Nc] -> bf16 transposed [Nc][K] ----------

__global__ void cvt_transpose_kernel(const float* __restrict__ in, bf16_t* __restrict__ outT,
                                     int K, int Nc) {
  int i = blockIdx.x * blockDim.x + threadIdx.x;
  if (i >= K * Nc) return;
  int k = i / Nc, n = i % Nc;
  outT[(size_t)n * K + k] = f2bf(in[i]);
}

// ---------- K1: FFN + residual + LayerNorm -> v  (16 nodes per block, 8 waves) ----------

__global__ void ffn_ln_kernel(const float* __restrict__ h,
                              const bf16_t* __restrict__ w1T, const float* __restrict__ b1,
                              const bf16_t* __restrict__ w2T, const float* __restrict__ b2,
                              const float* __restrict__ ln_g, const float* __restrict__ ln_b,
                              float* __restrict__ v_out, int N) {
  __shared__ bf16_t Asm[16 * DD];     // h tile, bf16
  __shared__ bf16_t Hsm[16 * HIDD];   // relu hidden, bf16
  __shared__ float  S[16 * DD];       // pre-LN (h + u)
  __shared__ float  red1[16][16], red2[16][16];
  __shared__ float  mu_s[16], rs_s[16];

  const int tid  = threadIdx.x;
  const int wave = tid >> 5;
  const int lane = tid & 31;
  const int node0 = blockIdx.x * 16;

  // stage h tile -> bf16 LDS
#pragma unroll
  for (int j = 0; j < 8; ++j) {
    int idx = j * 256 + tid;                       // 2048 elements
    int row = idx >> 7;
    float val = (node0 + row < N) ? h[(size_t)node0 * DD + idx] : 0.f;
    Asm[idx] = f2bf(val);
  }
  __syncthreads();

  // GEMM1: [16x128] @ [128x512] -> relu -> Hsm.  4 K-steps; wave w owns N-tiles 4w..4w+3.
  v16bf afr[4];
#pragma unroll
  for (int ks = 0; ks < 4; ++ks) afr[ks] = load_A_frag(Asm, DD, ks * 32, lane);

#pragma unroll
  for (int t = 0; t < 4; ++t) {
    const int ntile = wave * 4 + t;
    v8f acc = {};
#pragma unroll
    for (int ks = 0; ks < 4; ++ks) {
      v16bf bfr = load_B_frag(w1T, DD, ntile * 16, ks * 32, lane);
      acc = wmma_bf16(afr[ks], bfr, acc);
    }
    const int col = ntile * 16 + (lane & 15);
    const float bias = b1[col];
#pragma unroll
    for (int i = 0; i < 8; ++i) {
      int m = i + 8 * (lane >> 4);
      float r = acc[i] + bias;
      Hsm[m * HIDD + col] = f2bf(r > 0.f ? r : 0.f);
    }
  }
  __syncthreads();

  // GEMM2: [16x512] @ [512x128] + b2 + residual -> S.  16 K-steps; wave w owns N-tile w.
  {
    const int ntile = wave;
    v8f acc = {};
#pragma unroll
    for (int ks = 0; ks < 16; ++ks) {
      v16bf a = load_A_frag(Hsm, HIDD, ks * 32, lane);
      v16bf b = load_B_frag(w2T, HIDD, ntile * 16, ks * 32, lane);
      acc = wmma_bf16(a, b, acc);
    }
    const int col = ntile * 16 + (lane & 15);
    const float bias = b2[col];
#pragma unroll
    for (int i = 0; i < 8; ++i) {
      int m = i + 8 * (lane >> 4);
      float res = (node0 + m < N) ? h[(size_t)(node0 + m) * DD + col] : 0.f;
      S[m * DD + col] = acc[i] + bias + res;
    }
  }
  __syncthreads();

  // LayerNorm over 128 per row: 16 threads per row, 8 cols each.
  {
    const int row = tid >> 4, part = tid & 15;
    float s = 0.f, s2 = 0.f;
#pragma unroll
    for (int j = 0; j < 8; ++j) {
      float x = S[row * DD + part * 8 + j];
      s += x; s2 += x * x;
    }
    red1[row][part] = s; red2[row][part] = s2;
  }
  __syncthreads();
  if (tid < 16) {
    float s = 0.f, s2 = 0.f;
#pragma unroll
    for (int p = 0; p < 16; ++p) { s += red1[tid][p]; s2 += red2[tid][p]; }
    float mu = s * (1.f / DD);
    float var = s2 * (1.f / DD) - mu * mu;
    mu_s[tid] = mu; rs_s[tid] = rsqrtf(var + 1e-5f);
  }
  __syncthreads();
  {
    const int row = tid >> 4, part = tid & 15;
    const float mu = mu_s[row], rs = rs_s[row];
    if (node0 + row < N) {
#pragma unroll
      for (int j = 0; j < 8; ++j) {
        int col = part * 8 + j;
        float x = S[row * DD + col];
        v_out[(size_t)(node0 + row) * DD + col] = (x - mu) * rs * ln_g[col] + ln_b[col];
      }
    }
  }
}

// ---------- K2: xl = v@Wl + bl, xr = v@Wr + br  (16 nodes per block) ----------

__global__ void proj_kernel(const float* __restrict__ v,
                            const bf16_t* __restrict__ WlT, const float* __restrict__ bl,
                            const bf16_t* __restrict__ WrT, const float* __restrict__ br,
                            float* __restrict__ xl, float* __restrict__ xr, int N) {
  __shared__ bf16_t Asm[16 * DD];
  const int tid  = threadIdx.x;
  const int wave = tid >> 5;
  const int lane = tid & 31;
  const int node0 = blockIdx.x * 16;

#pragma unroll
  for (int j = 0; j < 8; ++j) {
    int idx = j * 256 + tid;
    int row = idx >> 7;
    float val = (node0 + row < N) ? v[(size_t)node0 * DD + idx] : 0.f;
    Asm[idx] = f2bf(val);
  }
  __syncthreads();

  v16bf afr[4];
#pragma unroll
  for (int ks = 0; ks < 4; ++ks) afr[ks] = load_A_frag(Asm, DD, ks * 32, lane);

  const int ntile = wave;                 // 8 waves, 8 N-tiles, both outputs per wave
  const int col = ntile * 16 + (lane & 15);

  v8f accL = {}, accR = {};
#pragma unroll
  for (int ks = 0; ks < 4; ++ks) {
    v16bf bL = load_B_frag(WlT, DD, ntile * 16, ks * 32, lane);
    accL = wmma_bf16(afr[ks], bL, accL);
  }
#pragma unroll
  for (int ks = 0; ks < 4; ++ks) {
    v16bf bR = load_B_frag(WrT, DD, ntile * 16, ks * 32, lane);
    accR = wmma_bf16(afr[ks], bR, accR);
  }
  const float biasL = bl[col], biasR = br[col];
#pragma unroll
  for (int i = 0; i < 8; ++i) {
    int m = i + 8 * (lane >> 4);
    if (node0 + m < N) {
      xl[(size_t)(node0 + m) * DD + col] = accL[i] + biasL;
      xr[(size_t)(node0 + m) * DD + col] = accR[i] + biasR;
    }
  }
}

// ---------- K3: xe WMMA + gather + leaky_relu + att-dot -> logits [E,H] ----------

__global__ void edge_logits_kernel(const float* __restrict__ edge_attr,
                                   const bf16_t* __restrict__ WeT,
                                   const int* __restrict__ src, const int* __restrict__ dst,
                                   const float* __restrict__ xl, const float* __restrict__ xr,
                                   const float* __restrict__ att,
                                   float* __restrict__ logits, int E) {
  __shared__ bf16_t Asm[16 * EDD];
  const int tid  = threadIdx.x;
  const int head = tid >> 5;              // wave == head (8 waves)
  const int lane = tid & 31;
  const int e0 = blockIdx.x * 16;

#pragma unroll
  for (int j = 0; j < 2; ++j) {
    int idx = j * 256 + tid;              // 512 elements
    int row = idx >> 5;
    float val = (e0 + row < E) ? edge_attr[(size_t)e0 * EDD + idx] : 0.f;
    Asm[idx] = f2bf(val);
  }
  __syncthreads();

  v16bf a = load_A_frag(Asm, EDD, 0, lane);
  v16bf b = load_B_frag(WeT, EDD, head * 16, 0, lane);
  v8f acc = {};
  acc = wmma_bf16(a, b, acc);             // xe for this head, 16 edges x 16 channels

  const int c   = lane & 15;
  const int col = head * CC + c;          // flat H*C column; also att index
  const float av = att[col];

  float t[8];
#pragma unroll
  for (int i = 0; i < 8; ++i) {
    int m = i + 8 * (lane >> 4);
    int e = e0 + m;
    float x = 0.f;
    if (e < E) {
      int s = src[e], d = dst[e];
      x = acc[i] + xl[(size_t)s * DD + col] + xr[(size_t)d * DD + col];
      x = (x > 0.f) ? x : 0.2f * x;       // leaky_relu(0.2)
    }
    t[i] = x * av;
  }
  // reduce over the 16 channels (lanes within each 16-lane group)
#pragma unroll
  for (int off = 8; off >= 1; off >>= 1) {
#pragma unroll
    for (int i = 0; i < 8; ++i) t[i] += __shfl_xor(t[i], off, 16);
  }
  if ((lane & 15) == 0) {
#pragma unroll
    for (int i = 0; i < 8; ++i) {
      int e = e0 + i + 8 * (lane >> 4);
      if (e < E) logits[(size_t)e * HH + head] = t[i];
    }
  }
}

// ---------- K4: segment max over dst (monotonic uint encoding of float) ----------

__device__ __forceinline__ unsigned enc_f32(float f) {
  unsigned b = __float_as_uint(f);
  return (b & 0x80000000u) ? ~b : (b | 0x80000000u);
}
__device__ __forceinline__ float dec_f32(unsigned k) {
  unsigned b = (k & 0x80000000u) ? (k ^ 0x80000000u) : ~k;
  return __uint_as_float(b);
}

__global__ void seg_max_kernel(const float* __restrict__ logits, const int* __restrict__ dst,
                               unsigned* __restrict__ nmax, int EH) {
  int i = blockIdx.x * blockDim.x + threadIdx.x;
  if (i >= EH) return;
  int e = i >> 3, hh = i & 7;
  atomicMax(&nmax[(size_t)dst[e] * HH + hh], enc_f32(logits[i]));
}

// ---------- K5: ex = exp(logit - max[dst]); den += ex ----------

__global__ void seg_exp_kernel(float* __restrict__ logits, const int* __restrict__ dst,
                               const unsigned* __restrict__ nmax, float* __restrict__ den,
                               int EH) {
  int i = blockIdx.x * blockDim.x + threadIdx.x;
  if (i >= EH) return;
  int e = i >> 3, hh = i & 7;
  size_t ni = (size_t)dst[e] * HH + hh;
  float m = dec_f32(nmax[ni]);
  float ex = __expf(logits[i] - m);
  logits[i] = ex;                         // overwrite with ex
  atomicAdd(&den[ni], ex);
}

// ---------- K6: alpha + message scatter (atomicAdd into L2-resident accumulator) ----------

__global__ void alpha_scatter_kernel(const float* __restrict__ ex,
                                     const int* __restrict__ src, const int* __restrict__ dst,
                                     const float* __restrict__ den, const float* __restrict__ xl,
                                     float* __restrict__ alpha_out, float* __restrict__ accum,
                                     int EH) {
  int i = blockIdx.x * blockDim.x + threadIdx.x;
  if (i >= EH) return;
  int e = i >> 3, hh = i & 7;
  int d = dst[e];
  float a = ex[i] / den[(size_t)d * HH + hh];
  alpha_out[i] = a;
  size_t sb = (size_t)src[e] * DD + hh * CC;
  size_t db = (size_t)d      * DD + hh * CC;
#pragma unroll
  for (int c = 0; c < CC; ++c) atomicAdd(&accum[db + c], xl[sb + c] * a);
}

// ---------- K7: output = LayerNorm(accum + bias_out + v)  (one wave per node) ----------

__global__ void final_ln_kernel(const float* __restrict__ accum, const float* __restrict__ bias_out,
                                const float* __restrict__ v, const float* __restrict__ ln_g,
                                const float* __restrict__ ln_b, float* __restrict__ out, int N) {
  int w = (blockIdx.x * blockDim.x + threadIdx.x) >> 5;
  int lane = threadIdx.x & 31;
  if (w >= N) return;
  size_t base = (size_t)w * DD;
  float x[4], s = 0.f, s2 = 0.f;
#pragma unroll
  for (int j = 0; j < 4; ++j) {
    int col = lane * 4 + j;
    x[j] = accum[base + col] + bias_out[col] + v[base + col];
    s += x[j]; s2 += x[j] * x[j];
  }
#pragma unroll
  for (int off = 16; off >= 1; off >>= 1) {
    s  += __shfl_xor(s,  off, 32);
    s2 += __shfl_xor(s2, off, 32);
  }
  float mu = s * (1.f / DD);
  float var = s2 * (1.f / DD) - mu * mu;
  float rs = rsqrtf(var + 1e-5f);
#pragma unroll
  for (int j = 0; j < 4; ++j) {
    int col = lane * 4 + j;
    out[base + col] = (x[j] - mu) * rs * ln_g[col] + ln_b[col];
  }
}

// ---------- host launcher ----------

extern "C" void kernel_launch(void* const* d_in, const int* in_sizes, int n_in,
                              void* d_out, int out_size, void* d_ws, size_t ws_size,
                              hipStream_t stream) {
  const float* h         = (const float*)d_in[0];
  const int*   ei        = (const int*)  d_in[1];   // [2,E] (JAX default int32)
  const float* edge_attr = (const float*)d_in[2];
  const float* w1        = (const float*)d_in[3];
  const float* b1        = (const float*)d_in[4];
  const float* w2        = (const float*)d_in[5];
  const float* b2        = (const float*)d_in[6];
  const float* ln_g      = (const float*)d_in[7];
  const float* ln_b      = (const float*)d_in[8];
  const float* Wl        = (const float*)d_in[9];
  const float* bl        = (const float*)d_in[10];
  const float* Wr        = (const float*)d_in[11];
  const float* br        = (const float*)d_in[12];
  const float* We        = (const float*)d_in[13];
  const float* att       = (const float*)d_in[14];
  const float* bias_out  = (const float*)d_in[15];

  const int N = in_sizes[0] / DD;
  const int E = in_sizes[2] / EDD;
  const int EH = E * HH;

  const int* src = ei;
  const int* dstp = ei + E;

  float* out_nodes = (float*)d_out;               // [N, 128]
  float* out_alpha = (float*)d_out + (size_t)N * DD;  // [E, 8]

  // workspace carve-up (256B aligned slices)
  char* ws = (char*)d_ws;
  size_t off = 0;
  auto take = [&](size_t bytes) -> void* {
    void* p = ws + off;
    off = (off + bytes + 255) & ~(size_t)255;
    return p;
  };
  float*    v_buf  = (float*)   take((size_t)N * DD * 4);
  float*    xl_buf = (float*)   take((size_t)N * DD * 4);
  float*    xr_buf = (float*)   take((size_t)N * DD * 4);
  float*    lg_buf = (float*)   take((size_t)EH * 4);       // logits, then ex
  unsigned* nmax   = (unsigned*)take((size_t)N * HH * 4);
  float*    den    = (float*)   take((size_t)N * HH * 4);
  float*    accum  = (float*)   take((size_t)N * DD * 4);
  bf16_t*   w1T    = (bf16_t*)  take((size_t)DD * HIDD * 2);
  bf16_t*   w2T    = (bf16_t*)  take((size_t)HIDD * DD * 2);
  bf16_t*   WlT    = (bf16_t*)  take((size_t)DD * DD * 2);
  bf16_t*   WrT    = (bf16_t*)  take((size_t)DD * DD * 2);
  bf16_t*   WeT    = (bf16_t*)  take((size_t)EDD * DD * 2);
  (void)ws_size;

  // zero-init atomics targets
  hipMemsetAsync(nmax,  0, (size_t)N * HH * 4, stream);
  hipMemsetAsync(den,   0, (size_t)N * HH * 4, stream);
  hipMemsetAsync(accum, 0, (size_t)N * DD * 4, stream);

  // K0: weight convert+transpose to bf16 [N][K]
  cvt_transpose_kernel<<<(DD * HIDD + 255) / 256, 256, 0, stream>>>(w1, w1T, DD,   HIDD);
  cvt_transpose_kernel<<<(HIDD * DD + 255) / 256, 256, 0, stream>>>(w2, w2T, HIDD, DD);
  cvt_transpose_kernel<<<(DD * DD   + 255) / 256, 256, 0, stream>>>(Wl, WlT, DD,   DD);
  cvt_transpose_kernel<<<(DD * DD   + 255) / 256, 256, 0, stream>>>(Wr, WrT, DD,   DD);
  cvt_transpose_kernel<<<(EDD * DD  + 255) / 256, 256, 0, stream>>>(We, WeT, EDD,  DD);

  const int nodeTiles = (N + 15) / 16;
  const int edgeTiles = (E + 15) / 16;

  // K1: FFN + residual + LN
  ffn_ln_kernel<<<nodeTiles, 256, 0, stream>>>(h, w1T, b1, w2T, b2, ln_g, ln_b, v_buf, N);
  // K2: xl / xr projections
  proj_kernel<<<nodeTiles, 256, 0, stream>>>(v_buf, WlT, bl, WrT, br, xl_buf, xr_buf, N);
  // K3: edge logits
  edge_logits_kernel<<<edgeTiles, 256, 0, stream>>>(edge_attr, WeT, src, dstp,
                                                    xl_buf, xr_buf, att, lg_buf, E);
  // K4-K6: segment softmax + message scatter
  seg_max_kernel<<<(EH + 255) / 256, 256, 0, stream>>>(lg_buf, dstp, nmax, EH);
  seg_exp_kernel<<<(EH + 255) / 256, 256, 0, stream>>>(lg_buf, dstp, nmax, den, EH);
  alpha_scatter_kernel<<<(EH + 255) / 256, 256, 0, stream>>>(lg_buf, src, dstp, den, xl_buf,
                                                             out_alpha, accum, EH);
  // K7: final LayerNorm
  final_ln_kernel<<<(N * 32 + 255) / 256, 256, 0, stream>>>(accum, bias_out, v_buf,
                                                            ln_g, ln_b, out_nodes, N);
}
